// DischargeNotesEncoder_17540646437712
// MI455X (gfx1250) — compile-verified, matrix operand order checked by reading, not run on hardware
//
#include <hip/hip_runtime.h>
#include <hip/hip_bf16.h>
#include <cstdint>

// Problem constants (match reference)
#define B_   2
#define L_   4096
#define H_   768
#define NH_  12
#define DH_  64
#define NL_  2
#define FF_  3072
#define OUT_ 384
#define CW_  128
#define NC_  (L_ / CW_)
#define EPS_ 1e-5f

typedef __attribute__((ext_vector_type(16))) _Float16 v16h;
typedef __attribute__((ext_vector_type(4)))  _Float16 v4h;
typedef __attribute__((ext_vector_type(8)))  float    v8f;
typedef __attribute__((ext_vector_type(4)))  float    v4f;

// ---------------------------------------------------------------- WMMA helper
__device__ inline v8f wmma16(v16h a, v16h b, v8f c) {
  // D = A(16x32,f16) * B(32x16,f16) + C(16x16,f32)
  return __builtin_amdgcn_wmma_f32_16x16x32_f16(false, a, false, b, (short)0, c,
                                                false, false);
}

__device__ inline v4h cvt4(v4f x) {
  v4h h;
  h[0] = (_Float16)x[0]; h[1] = (_Float16)x[1];
  h[2] = (_Float16)x[2]; h[3] = (_Float16)x[3];
  return h;
}

// Fragment-linear LDS layout (CDNA5 ISA 7.12.2):
//   A frag (16x32 MxK): lane = 16*((k>>3)&1) + m ; elem = 2*((k>>4)*4+((k>>1)&3)) + (k&1)
//   B frag (32x16 KxN): lane = 16*((k>>3)&1) + n ; elem as above
// Each lane's 16 halves are contiguous (32B) -> fragment load = 2x ds_load_b128.
__device__ inline v16h frag_ld(const _Float16* p) {
  return *(const v16h*)p;
}

__device__ inline float gelu_f(float x) {
  const float c0 = 0.7978845608028654f, c1 = 0.044715f;
  float x3 = x * x * x;
  return 0.5f * x * (1.f + tanhf(c0 * (x + c1 * x3)));
}

__device__ inline float block_reduce_sum256(float v, float* red) {
  const int t = threadIdx.x;
  red[t] = v;
  __syncthreads();
#pragma unroll
  for (int o = 128; o > 0; o >>= 1) {
    if (t < o) red[t] += red[t + o];
    __syncthreads();
  }
  float r = red[0];
  __syncthreads();
  return r;
}

// --------------------------------------------------- GEMM: C = act(A*W + bias)
// A:[M,K] f32, W:[K,N] f32. Tile 128x128, 8 waves (2x4), K-step 32.
// Software pipeline: global loads for tile k+1 issued before WMMAs of tile k.
__global__ void __launch_bounds__(256)
gemm_bias_kernel(const float* __restrict__ A, const float* __restrict__ W,
                 const float* __restrict__ bias, float* __restrict__ C,
                 int M, int N, int K, int act) {
  __shared__ __align__(32) _Float16 As[8 * 32 * 16];  // 128x32 frag-linear
  __shared__ __align__(32) _Float16 Bs[8 * 32 * 16];  // 32x128 frag-linear
  const int m0 = blockIdx.y * 128, n0 = blockIdx.x * 128;
  const int t = threadIdx.x, wave = t >> 5, lane = t & 31;
  const int wr = wave >> 2, wc = wave & 3;

  v8f acc[4][2];
#pragma unroll
  for (int i = 0; i < 4; ++i)
#pragma unroll
    for (int j = 0; j < 2; ++j)
#pragma unroll
      for (int r = 0; r < 8; ++r) acc[i][j][r] = 0.f;

  v4f a4[4], w4[4];
  // prologue: batched loads of tile 0 (all in flight before first use)
#pragma unroll
  for (int i = 0; i < 4; ++i) {
    int q = t + i * 256;
    int row = q >> 3, kc = (q & 7) << 2;
    a4[i] = *(const v4f*)&A[(size_t)(m0 + row) * K + kc];
  }
#pragma unroll
  for (int i = 0; i < 4; ++i) {
    int q = t + i * 256;
    int r = q >> 5, nc4 = (q & 31) << 2;
    w4[i] = *(const v4f*)&W[(size_t)r * N + n0 + nc4];
  }

  for (int k0 = 0; k0 < K; k0 += 32) {
    // stage current tile regs -> LDS (single wait for the batched loads)
#pragma unroll
    for (int i = 0; i < 4; ++i) {
      int q = t + i * 256;
      int row = q >> 3, kc = (q & 7) << 2;
      int msub = row >> 4, m = row & 15;
      int flane = (((kc >> 3) & 1) << 4) + m;
      int eb = 2 * (((kc >> 4) << 2) + ((kc >> 1) & 3));
      *(v4h*)&As[((msub * 32 + flane) << 4) + eb] = cvt4(a4[i]);
    }
#pragma unroll
    for (int i = 0; i < 4; ++i) {
      int q = t + i * 256;
      int r = q >> 5, nc4 = (q & 31) << 2;
      int hs = (r >> 3) & 1;
      int elem = 2 * (((r >> 4) << 2) + ((r >> 1) & 3)) + (r & 1);
#pragma unroll
      for (int u = 0; u < 4; ++u) {
        int c = nc4 + u;
        Bs[(((c >> 4) * 32 + hs * 16 + (c & 15)) << 4) + elem] = (_Float16)w4[i][u];
      }
    }
    __syncthreads();
    // issue next-tile global loads; they complete during the WMMAs below
    const int k1 = k0 + 32;
    if (k1 < K) {
#pragma unroll
      for (int i = 0; i < 4; ++i) {
        int q = t + i * 256;
        int row = q >> 3, kc = (q & 7) << 2;
        a4[i] = *(const v4f*)&A[(size_t)(m0 + row) * K + k1 + kc];
      }
#pragma unroll
      for (int i = 0; i < 4; ++i) {
        int q = t + i * 256;
        int r = q >> 5, nc4 = (q & 31) << 2;
        w4[i] = *(const v4f*)&W[(size_t)(k1 + r) * N + n0 + nc4];
      }
      if (k1 + 32 < K)  // warm L2 two tiles ahead (global_prefetch_b8)
        __builtin_prefetch(&W[(size_t)(k1 + 32 + (t >> 3)) * N + n0 + ((t & 7) << 4)], 0, 1);
    }
    // compute on staged tile
    v16h af[4], bf[2];
#pragma unroll
    for (int i = 0; i < 4; ++i)
      af[i] = frag_ld(&As[((wr * 4 + i) * 32 + lane) << 4]);
#pragma unroll
    for (int j = 0; j < 2; ++j)
      bf[j] = frag_ld(&Bs[((wc * 2 + j) * 32 + lane) << 4]);
#pragma unroll
    for (int i = 0; i < 4; ++i)
#pragma unroll
      for (int j = 0; j < 2; ++j)
        acc[i][j] = wmma16(af[i], bf[j], acc[i][j]);
    __syncthreads();
  }
  const int mb = (lane >> 4) << 3, nc = lane & 15;
#pragma unroll
  for (int i = 0; i < 4; ++i)
#pragma unroll
    for (int j = 0; j < 2; ++j) {
      int n = n0 + (wc * 2 + j) * 16 + nc;
      float bv = bias[n];
#pragma unroll
      for (int r = 0; r < 8; ++r) {
        int m = m0 + (wr * 4 + i) * 16 + mb + r;
        float val = acc[i][j][r] + bv;
        if (act == 1) val = gelu_f(val);
        C[(size_t)m * N + n] = val;
      }
    }
}

// -------------------------------------------- Embedding + LayerNorm (per token)
__global__ void __launch_bounds__(256)
embed_ln_kernel(const int* __restrict__ ids, const float* __restrict__ wemb,
                const float* __restrict__ pemb, const float* __restrict__ sc,
                const float* __restrict__ bi, float* __restrict__ X) {
  __shared__ float red[256];
  const size_t tok = blockIdx.x;
  const int id = ids[tok];
  const int l = (int)(tok % L_);
  float v[3];
  float s = 0.f;
#pragma unroll
  for (int i = 0; i < 3; ++i) {
    int c = threadIdx.x + i * 256;
    v[i] = wemb[(size_t)id * H_ + c] + pemb[(size_t)l * H_ + c];
    s += v[i];
  }
  float mean = block_reduce_sum256(s, red) * (1.f / H_);
  float s2 = 0.f;
#pragma unroll
  for (int i = 0; i < 3; ++i) { float d = v[i] - mean; s2 += d * d; }
  float var = block_reduce_sum256(s2, red) * (1.f / H_);
  float rstd = rsqrtf(var + EPS_);
#pragma unroll
  for (int i = 0; i < 3; ++i) {
    int c = threadIdx.x + i * 256;
    X[tok * H_ + c] = (v[i] - mean) * rstd * sc[c] + bi[c];
  }
}

// ------------------------------------------------ X = LN(Xin + Yin) * s + b
__global__ void __launch_bounds__(256)
add_ln_kernel(const float* __restrict__ Xin, const float* __restrict__ Yin,
              const float* __restrict__ sc, const float* __restrict__ bi,
              float* __restrict__ Xout) {
  __shared__ float red[256];
  const size_t tok = blockIdx.x;
  float v[3];
  float s = 0.f;
#pragma unroll
  for (int i = 0; i < 3; ++i) {
    int c = threadIdx.x + i * 256;
    v[i] = Xin[tok * H_ + c] + Yin[tok * H_ + c];
    s += v[i];
  }
  float mean = block_reduce_sum256(s, red) * (1.f / H_);
  float s2 = 0.f;
#pragma unroll
  for (int i = 0; i < 3; ++i) { float d = v[i] - mean; s2 += d * d; }
  float var = block_reduce_sum256(s2, red) * (1.f / H_);
  float rstd = rsqrtf(var + EPS_);
#pragma unroll
  for (int i = 0; i < 3; ++i) {
    int c = threadIdx.x + i * 256;
    Xout[tok * H_ + c] = (v[i] - mean) * rstd * sc[c] + bi[c];
  }
}

// ------------------------- Banded local attention: 1 wave per (b,h,chunk,16q)
// Scores via WMMA (Q x K^T), masked softmax (+ global k0 column) writing f16
// probability fragments, then P x V via WMMA. Token-0 rows overwritten later.
__global__ void __launch_bounds__(32)
attn_local_kernel(const float* __restrict__ Qg, const float* __restrict__ Kg,
                  const float* __restrict__ Vg, const int* __restrict__ amask,
                  float* __restrict__ O) {
  const int lane = threadIdx.x;
  const int qt = blockIdx.x;            // 0..7 : 16-query tile within chunk
  const int n  = blockIdx.y;            // chunk 0..31
  const int bh = blockIdx.z;            // 0..B*NH-1
  const int b = bh / NH_, h = bh % NH_;
  const int qbase = n * CW_ + qt * 16;
  const float scale = 0.125f;           // 1/sqrt(64)
  const size_t rowb = (size_t)b * L_;

  __shared__ __align__(32) _Float16 Qf[2 * 32 * 16];    // Q A-fragments
  __shared__ __align__(32) _Float16 KVf[4 * 32 * 16];   // K (score) / V (PV) B-frags
  __shared__ __align__(32) _Float16 Pf[12 * 32 * 16];   // probability A-fragments
  __shared__ float S[16][388];                          // raw scores (f32)
  __shared__ float pg[16];                              // global-key probability

  // stage Q tile (16x64) into two A-fragments: batched loads, packed b64 stores
  v4f qreg[8];
#pragma unroll
  for (int i = 0; i < 8; ++i) {
    int q = lane + i * 32;
    int row = q >> 4, kc = (q & 15) << 2;
    qreg[i] = *(const v4f*)&Qg[(rowb + qbase + row) * H_ + h * 64 + kc];
  }
#pragma unroll
  for (int i = 0; i < 8; ++i) {
    int q = lane + i * 32;
    int row = q >> 4, kc = (q & 15) << 2;
    int g = kc >> 5, kl = kc & 31;
    int flane = (((kl >> 3) & 1) << 4) + row;
    int eb = 2 * (((kl >> 4) << 2) + ((kl >> 1) & 3));
    *(v4h*)&Qf[((g * 32 + flane) << 4) + eb] = cvt4(qreg[i]);
  }
  __syncthreads();
  v16h qa0 = frag_ld(&Qf[(0 * 32 + lane) << 4]);
  v16h qa1 = frag_ld(&Qf[(1 * 32 + lane) << 4]);
  const int kstart = (n - 1) * CW_;     // position of local key column 0

  // ---- score phase: 12 groups of 32 keys
  for (int kt = 0; kt < 12; ++kt) {
    v4f kreg[16];
#pragma unroll
    for (int i = 0; i < 16; ++i) {      // batched loads (before barrier)
      int q = lane + i * 32;
      int r = q >> 4, kc = (q & 15) << 2;
      int pos = kstart + kt * 32 + r;
      v4f z = {0.f, 0.f, 0.f, 0.f};
      kreg[i] = z;
      if (pos >= 0 && pos < L_)
        kreg[i] = *(const v4f*)&Kg[(rowb + pos) * H_ + h * 64 + kc];
    }
    __syncthreads();
#pragma unroll
    for (int i = 0; i < 16; ++i) {      // packed stores into B-frag layout
      int q = lane + i * 32;
      int r = q >> 4, kc = (q & 15) << 2;
      int sub = r >> 4, nn = r & 15;
      int g = kc >> 5, kl = kc & 31;
      int flane = (((kl >> 3) & 1) << 4) + nn;
      int eb = 2 * (((kl >> 4) << 2) + ((kl >> 1) & 3));
      *(v4h*)&KVf[(((sub * 2 + g) * 32 + flane) << 4) + eb] = cvt4(kreg[i]);
    }
    __syncthreads();
#pragma unroll
    for (int sub = 0; sub < 2; ++sub) {
      v8f sacc;
#pragma unroll
      for (int r = 0; r < 8; ++r) sacc[r] = 0.f;
      v16h kb0 = frag_ld(&KVf[((sub * 2 + 0) * 32 + lane) << 4]);
      v16h kb1 = frag_ld(&KVf[((sub * 2 + 1) * 32 + lane) << 4]);
      sacc = wmma16(qa0, kb0, sacc);
      sacc = wmma16(qa1, kb1, sacc);
      const int mb = (lane >> 4) << 3, nc = lane & 15;
#pragma unroll
      for (int r = 0; r < 8; ++r)
        S[mb + r][kt * 32 + sub * 16 + nc] = sacc[r] * scale;
    }
  }
  __syncthreads();

  // ---- masked softmax (each of 16 lanes owns one query row)
  if (lane < 16) {
    const int row = lane;
    const int pos_q = qbase + row;
    float sg = 0.f;                     // global key (token 0)
    for (int d = 0; d < 64; ++d)
      sg += (float)Qf[(((d >> 5) * 32 + (((d >> 3) & 1) << 4) + row) << 4) +
                      2 * ((((d & 31) >> 4) << 2) + ((d >> 1) & 3)) + (d & 1)] *
            Kg[rowb * H_ + h * 64 + d];
    sg *= scale;
    float mx = sg;
    for (int j = 0; j < 384; ++j) {
      int pos = kstart + j;
      int pc = pos < 0 ? 0 : (pos >= L_ ? L_ - 1 : pos);
      int diff = pos_q - pos; if (diff < 0) diff = -diff;
      bool ok = (pos >= 0) && (pos < L_) && (pos != 0) && (diff <= CW_) &&
                (amask[b * L_ + pc] > 0);
      float sv = ok ? S[row][j] : -1.0e9f;
      S[row][j] = sv;
      mx = fmaxf(mx, sv);
    }
    float sum = 0.f;
    for (int j = 0; j < 384; ++j) {
      float e = __expf(S[row][j] - mx);
      S[row][j] = e; sum += e;
    }
    float eg = __expf(sg - mx);
    sum += eg;
    float inv = 1.f / sum;
    // write normalized probabilities directly as f16 A-fragments
    for (int j = 0; j < 384; ++j) {
      int kt = j >> 5, kl = j & 31;
      int flane = (((kl >> 3) & 1) << 4) + row;
      int elem = 2 * (((kl >> 4) << 2) + ((kl >> 1) & 3)) + (kl & 1);
      Pf[((kt * 32 + flane) << 4) + elem] = (_Float16)(S[row][j] * inv);
    }
    pg[row] = eg * inv;
  }

  // ---- P x V phase
  v8f oacc[4];
#pragma unroll
  for (int j = 0; j < 4; ++j)
#pragma unroll
    for (int r = 0; r < 8; ++r) oacc[j][r] = 0.f;

  for (int kt = 0; kt < 12; ++kt) {
    v4f vreg[16];
#pragma unroll
    for (int i = 0; i < 16; ++i) {      // batched loads (before barrier)
      int q = lane + i * 32;
      int r = q >> 4, nc4 = (q & 15) << 2;
      int pos = kstart + kt * 32 + r;
      v4f z = {0.f, 0.f, 0.f, 0.f};
      vreg[i] = z;
      if (pos >= 0 && pos < L_)
        vreg[i] = *(const v4f*)&Vg[(rowb + pos) * H_ + h * 64 + nc4];
    }
    __syncthreads();
#pragma unroll
    for (int i = 0; i < 16; ++i) {      // scatter into B-frag layout (k=key)
      int q = lane + i * 32;
      int r = q >> 4, nc4 = (q & 15) << 2;
      int hs = (r >> 3) & 1;
      int elem = 2 * (((r >> 4) << 2) + ((r >> 1) & 3)) + (r & 1);
#pragma unroll
      for (int u = 0; u < 4; ++u) {
        int c = nc4 + u;
        KVf[(((c >> 4) * 32 + hs * 16 + (c & 15)) << 4) + elem] = (_Float16)vreg[i][u];
      }
    }
    __syncthreads();
    v16h pa = frag_ld(&Pf[(kt * 32 + lane) << 4]);
#pragma unroll
    for (int j = 0; j < 4; ++j) {
      v16h vb = frag_ld(&KVf[(j * 32 + lane) << 4]);
      oacc[j] = wmma16(pa, vb, oacc[j]);
    }
  }
  const int mb = (lane >> 4) << 3, nc = lane & 15;
#pragma unroll
  for (int j = 0; j < 4; ++j) {
    int d = j * 16 + nc;
    float v0 = Vg[rowb * H_ + h * 64 + d];
#pragma unroll
    for (int r = 0; r < 8; ++r) {
      int m = mb + r;
      O[(rowb + qbase + m) * H_ + h * 64 + d] = oacc[j][r] + pg[m] * v0;
    }
  }
}

// ----------------------- full-sequence attention for the global token (q = 0)
__global__ void __launch_bounds__(256)
attn_global_q0_kernel(const float* __restrict__ Qg, const float* __restrict__ Kg,
                      const float* __restrict__ Vg, const int* __restrict__ amask,
                      float* __restrict__ O) {
  const int bh = blockIdx.x;
  const int b = bh / NH_, h = bh % NH_;
  const int t = threadIdx.x;
  const size_t rowb = (size_t)b * L_;
  __shared__ float sbuf[L_];
  __shared__ float red[256];
  __shared__ float qrow[64];
  if (t < 64) qrow[t] = Qg[rowb * H_ + h * 64 + t];
  __syncthreads();
  float lmax = -1.0e30f;
  for (int i = 0; i < L_ / 256; ++i) {
    int l = t + i * 256;
    float s = 0.f;
    for (int d = 0; d < 64; ++d)
      s += qrow[d] * Kg[(rowb + l) * H_ + h * 64 + d];
    s *= 0.125f;
    if (amask[b * L_ + l] <= 0) s = -1.0e9f;
    sbuf[l] = s;
    lmax = fmaxf(lmax, s);
  }
  red[t] = lmax; __syncthreads();
#pragma unroll
  for (int o = 128; o > 0; o >>= 1) {
    if (t < o) red[t] = fmaxf(red[t], red[t + o]);
    __syncthreads();
  }
  float mx = red[0]; __syncthreads();
  float lsum = 0.f;
  for (int i = 0; i < L_ / 256; ++i) {
    int l = t + i * 256;
    float e = __expf(sbuf[l] - mx);
    sbuf[l] = e; lsum += e;
  }
  red[t] = lsum; __syncthreads();
#pragma unroll
  for (int o = 128; o > 0; o >>= 1) {
    if (t < o) red[t] += red[t + o];
    __syncthreads();
  }
  float inv = 1.f / red[0]; __syncthreads();
  if (t < 64) {
    float acc = 0.f;
    for (int l = 0; l < L_; ++l)
      acc += sbuf[l] * Vg[(rowb + l) * H_ + h * 64 + t];
    O[rowb * H_ + h * 64 + t] = acc * inv;
  }
}

// ------------------------------------------------------------------ cls copy
__global__ void cls_copy_kernel(float* __restrict__ out) {
  int i = threadIdx.x + blockIdx.x * blockDim.x;
  if (i < B_ * OUT_) {
    int b = i / OUT_, o = i % OUT_;
    out[(size_t)B_ * L_ * OUT_ + i] = out[(size_t)b * L_ * OUT_ + o];
  }
}

// =========================================================== host entry point
extern "C" void kernel_launch(void* const* d_in, const int* in_sizes, int n_in,
                              void* d_out, int out_size, void* d_ws, size_t ws_size,
                              hipStream_t stream) {
  (void)in_sizes; (void)n_in; (void)out_size; (void)ws_size;
  const int*   ids   = (const int*)d_in[0];
  const int*   amask = (const int*)d_in[1];
  const float* wemb  = (const float*)d_in[2];
  const float* pemb  = (const float*)d_in[3];
  const float* eln_s = (const float*)d_in[4];
  const float* eln_b = (const float*)d_in[5];
  const float* Wq  = (const float*)d_in[6];
  const float* bq  = (const float*)d_in[7];
  const float* Wk  = (const float*)d_in[8];
  const float* bk  = (const float*)d_in[9];
  const float* Wv  = (const float*)d_in[10];
  const float* bv  = (const float*)d_in[11];
  const float* Wo  = (const float*)d_in[12];
  const float* bo  = (const float*)d_in[13];
  const float* l1s = (const float*)d_in[14];
  const float* l1b = (const float*)d_in[15];
  const float* W1  = (const float*)d_in[16];
  const float* b1  = (const float*)d_in[17];
  const float* W2  = (const float*)d_in[18];
  const float* b2  = (const float*)d_in[19];
  const float* l2s = (const float*)d_in[20];
  const float* l2b = (const float*)d_in[21];
  const float* Wfc = (const float*)d_in[22];
  const float* bfc = (const float*)d_in[23];

  float* out = (float*)d_out;
  float* ws  = (float*)d_ws;
  const size_t BLH = (size_t)B_ * L_ * H_;
  float* X  = ws;
  float* Qb = ws + 1 * BLH;
  float* Kb = ws + 2 * BLH;
  float* Vb = ws + 3 * BLH;
  float* T1 = ws + 4 * BLH;
  float* T0 = ws + 5 * BLH;            // B*L*FF floats

  const int M = B_ * L_;
  dim3 gH(H_ / 128, M / 128), gF(FF_ / 128, M / 128), gO(OUT_ / 128, M / 128);
  dim3 gA(CW_ / 16, NC_, B_ * NH_);

  embed_ln_kernel<<<M, 256, 0, stream>>>(ids, wemb, pemb, eln_s, eln_b, X);

  for (int l = 0; l < NL_; ++l) {
    const size_t wHH = (size_t)l * H_ * H_;
    gemm_bias_kernel<<<gH, 256, 0, stream>>>(X, Wq + wHH, bq + l * H_, Qb, M, H_, H_, 0);
    gemm_bias_kernel<<<gH, 256, 0, stream>>>(X, Wk + wHH, bk + l * H_, Kb, M, H_, H_, 0);
    gemm_bias_kernel<<<gH, 256, 0, stream>>>(X, Wv + wHH, bv + l * H_, Vb, M, H_, H_, 0);
    attn_local_kernel<<<gA, 32, 0, stream>>>(Qb, Kb, Vb, amask, T1);
    attn_global_q0_kernel<<<B_ * NH_, 256, 0, stream>>>(Qb, Kb, Vb, amask, T1);
    gemm_bias_kernel<<<gH, 256, 0, stream>>>(T1, Wo + wHH, bo + l * H_, T0, M, H_, H_, 0);
    add_ln_kernel<<<M, 256, 0, stream>>>(X, T0, l1s + l * H_, l1b + l * H_, X);
    gemm_bias_kernel<<<gF, 256, 0, stream>>>(X, W1 + (size_t)l * H_ * FF_, b1 + l * FF_,
                                             T0, M, FF_, H_, 1);
    gemm_bias_kernel<<<gH, 256, 0, stream>>>(T0, W2 + (size_t)l * FF_ * H_, b2 + l * H_,
                                             T1, M, H_, FF_, 0);
    add_ln_kernel<<<M, 256, 0, stream>>>(X, T1, l2s + l * H_, l2b + l * H_, X);
  }

  gemm_bias_kernel<<<gO, 256, 0, stream>>>(X, Wfc, bfc, out, M, OUT_, H_, 0);
  cls_copy_kernel<<<(B_ * OUT_ + 255) / 256, 256, 0, stream>>>(out);
}
